// GCNClassification_84035330113566
// MI455X (gfx1250) — matrix operand order repaired and checked by reading, not compiled, not run on hardware
//
#include <hip/hip_runtime.h>

// GCN 3-layer + mean-pool + linear head for MI455X (gfx1250, wave32).
// GEMMs use v_wmma_f32_16x16x32_f16; edge aggregation uses f32 global atomics.

#define HID 128
#define NGRAPHS 512
#define NCLS 10

typedef __attribute__((ext_vector_type(16))) _Float16 v16h;
typedef __attribute__((ext_vector_type(8)))  _Float16 v8h;
typedef __attribute__((ext_vector_type(4)))  _Float16 v4h;
typedef __attribute__((ext_vector_type(8)))  float    v8f;

// ---------------- setup kernels ----------------

// deg[i] = 1.0 (self loop)
__global__ void k_fill1(float* __restrict__ deg, int n) {
  int i = blockIdx.x * blockDim.x + threadIdx.x;
  if (i < n) deg[i] = 1.0f;
}

// deg[dst[e]] += 1
__global__ void k_deg(const int* __restrict__ dst, float* __restrict__ deg, int nE) {
  int e = blockIdx.x * blockDim.x + threadIdx.x;
  if (e < nE) atomicAdd(&deg[dst[e]], 1.0f);
}

// dinv[i] = rsqrt(deg[i])  (deg >= 1 always, self-loop)
__global__ void k_rsqrt(float* __restrict__ d, int n) {
  int i = blockIdx.x * blockDim.x + threadIdx.x;
  if (i < n) d[i] = rsqrtf(d[i]);
}

// s[i] = dinv[i]^2  (self-loop norm contribution)
__global__ void k_selfnorm(const float* __restrict__ dinv, float* __restrict__ s, int n) {
  int i = blockIdx.x * blockDim.x + threadIdx.x;
  if (i < n) { float v = dinv[i]; s[i] = v * v; }
}

// norm[e] = dinv[src]*dinv[dst];  s[dst] += norm[e]
__global__ void k_edgenorm(const int* __restrict__ src, const int* __restrict__ dst,
                           const float* __restrict__ dinv, float* __restrict__ norm,
                           float* __restrict__ s, int nE) {
  int e = blockIdx.x * blockDim.x + threadIdx.x;
  if (e >= nE) return;
  float w = dinv[src[e]] * dinv[dst[e]];
  norm[e] = w;
  atomicAdd(&s[dst[e]], w);
}

// Layer 0 collapses: x1[i,f] = relu(s[i]*W0[f] + b0[f])  -> f16 for WMMA
// one thread = 4 consecutive feats of one node (b128 loads, b64 stores)
__global__ void k_layer0(const float* __restrict__ s, const float* __restrict__ W0,
                         const float* __restrict__ b0, _Float16* __restrict__ x16,
                         int nquad) {
  int q = blockIdx.x * blockDim.x + threadIdx.x;
  if (q >= nquad) return;
  int f4 = (q & 31);                // feat quad 0..31
  float sv = s[q >> 5];
  float4 w = ((const float4*)W0)[f4];
  float4 b = ((const float4*)b0)[f4];
  v4h o;
  o[0] = (_Float16)fmaxf(fmaf(sv, w.x, b.x), 0.0f);
  o[1] = (_Float16)fmaxf(fmaf(sv, w.y, b.y), 0.0f);
  o[2] = (_Float16)fmaxf(fmaf(sv, w.z, b.z), 0.0f);
  o[3] = (_Float16)fmaxf(fmaf(sv, w.w, b.w), 0.0f);
  *(v4h*)(x16 + (size_t)q * 4) = o;
}

// Wt16[n*128+k] = (f16) W[k*128+n]  (transpose so B-lane reads are contiguous in K)
__global__ void k_prepw(const float* __restrict__ W, _Float16* __restrict__ Wt) {
  int t = blockIdx.x * blockDim.x + threadIdx.x;
  if (t >= HID * HID) return;
  int n = t >> 7, k = t & (HID - 1);
  Wt[t] = (_Float16)W[k * HID + n];
}

// ---------------- WMMA GEMM: H[N x 128] = X[N x 128] @ W[128 x 128] ----------------
// Block = 256 threads = 8 waves; block owns one FULL 16-row tile (no tail handling
// here -- host launches nrows/16 blocks; remainder rows go to k_gemm_tail).
__global__ void __launch_bounds__(256) k_gemm(const _Float16* __restrict__ X,
                                              const _Float16* __restrict__ Wt,
                                              float* __restrict__ H) {
  __shared__ _Float16 sA[16 * HID];    // x tile, row-major [m][k]   (4 KB)
  __shared__ _Float16 sB[HID * HID];   // Wt,     [n][k]             (32 KB)
  const int tid  = threadIdx.x;
  const int row0 = blockIdx.x * 16;

  // stage A: 2048 halfs, 256 threads x 8 halfs (b128 per thread)
  {
    int idx = tid * 8;
    *(v8h*)(sA + idx) = *(const v8h*)(X + (size_t)row0 * HID + idx);
  }
  // stage B: 16384 halfs, 8 passes of 256 x b128
#pragma unroll
  for (int it = 0; it < 8; ++it) {
    int idx = (it * 256 + tid) * 8;
    *(v8h*)(sB + idx) = *(const v8h*)(Wt + idx);
  }
  __syncthreads();

  const int wave  = tid >> 5;       // col tile 0..7
  const int lane  = tid & 31;
  const int half  = lane >> 4;      // lane group
  const int idx16 = lane & 15;
  const int m = idx16;              // A row within tile (lanes repeat per half)
  const int n = wave * 16 + idx16;  // global col

  v8f acc = {};
#pragma unroll
  for (int ks = 0; ks < 4; ++ks) {
    // A operand, this code's contiguous k order: lane half h -> K {kb..kb+15}, kb = ks*32 + half*16
    int kb = ks * 32 + half * 16;
    v8h a0 = *(const v8h*)(sA + m * HID + kb);
    v8h a1 = *(const v8h*)(sA + m * HID + kb + 8);
    // B operand, same contiguous k order as A: lanes h=0 -> K = ks*32 + 0..15 ; h=1 -> +16
    v16h b = *(const v16h*)(sB + n * HID + ks * 32 + half * 16);
    v16h a;
#pragma unroll
    for (int j = 0; j < 8; ++j) { a[j] = a0[j]; a[j + 8] = a1[j]; }
    acc = __builtin_amdgcn_wmma_f32_16x16x32_f16(
        /*neg_a=*/false, a, /*neg_b=*/false, b,
        /*c_mod=*/(short)0, acc, /*reuse_a=*/false, /*reuse_b=*/false);
  }
  // C/D layout: VGPR r -> row = r + half*8, col = idx16; unconditional stores
  float* hp = H + (size_t)(row0 + half * 8) * HID + n;
#pragma unroll
  for (int r = 0; r < 8; ++r) hp[(size_t)r * HID] = acc[r];
}

// scalar fallback for the (nrows % 16) remainder rows; not launched when N%16==0
__global__ void k_gemm_tail(const _Float16* __restrict__ X, const _Float16* __restrict__ Wt,
                            float* __restrict__ H, int rowBeg, int nrows) {
  int t = blockIdx.x * blockDim.x + threadIdx.x;
  int row = rowBeg + (t >> 7);
  int n = t & (HID - 1);
  if (row >= nrows) return;
  float acc = 0.0f;
  for (int k = 0; k < HID; ++k)
    acc += (float)X[(size_t)row * HID + k] * (float)Wt[n * HID + k];
  H[(size_t)row * HID + n] = acc;
}

// ---------------- aggregation ----------------

// agg[i,f] = dinv[i]^2 * h[i,f]   (self loop); one thread = 4 feats (b128 in/out)
__global__ void k_agginit(const float* __restrict__ H, const float* __restrict__ dinv,
                          float* __restrict__ agg, int nquad) {
  int q = blockIdx.x * blockDim.x + threadIdx.x;
  if (q >= nquad) return;
  float di = dinv[q >> 5];
  float d2 = di * di;
  float4 v = ((const float4*)H)[q];
  float4 o = make_float4(d2 * v.x, d2 * v.y, d2 * v.z, d2 * v.w);
  ((float4*)agg)[q] = o;
}

// one wave per edge; lane moves 4 feats: agg[dst] += norm * h[src]
__global__ void __launch_bounds__(256) k_scatter(const int* __restrict__ src,
                                                 const int* __restrict__ dst,
                                                 const float* __restrict__ norm,
                                                 const float* __restrict__ H,
                                                 float* __restrict__ agg, int nE) {
  int e = blockIdx.x * 8 + (threadIdx.x >> 5);
  if (e >= nE) return;
  int lane = threadIdx.x & 31;
  int s = src[e], d = dst[e];
  float w = norm[e];
  float4 v = *((const float4*)(H + (size_t)s * HID) + lane);
  float* ap = agg + (size_t)d * HID + lane * 4;
  atomicAdd(ap + 0, w * v.x);
  atomicAdd(ap + 1, w * v.y);
  atomicAdd(ap + 2, w * v.z);
  atomicAdd(ap + 3, w * v.w);
}

// x_next = relu(agg + b); f16 always (next GEMM), f32 too on last layer (pooling)
// one thread = 4 feats
__global__ void k_finish(const float* __restrict__ agg, const float* __restrict__ b,
                         _Float16* __restrict__ x16, float* __restrict__ x32, int nquad) {
  int q = blockIdx.x * blockDim.x + threadIdx.x;
  if (q >= nquad) return;
  float4 a  = ((const float4*)agg)[q];
  float4 bb = ((const float4*)b)[q & 31];
  float4 o = make_float4(fmaxf(a.x + bb.x, 0.0f), fmaxf(a.y + bb.y, 0.0f),
                         fmaxf(a.z + bb.z, 0.0f), fmaxf(a.w + bb.w, 0.0f));
  v4h h16 = { (_Float16)o.x, (_Float16)o.y, (_Float16)o.z, (_Float16)o.w };
  *(v4h*)(x16 + (size_t)q * 4) = h16;
  if (x32) ((float4*)x32)[q] = o;
}

// ---------------- pooling + head ----------------

__global__ void k_zeropool(float* __restrict__ pooled, float* __restrict__ cnt) {
  int t = blockIdx.x * blockDim.x + threadIdx.x;
  if (t < NGRAPHS * HID) pooled[t] = 0.0f;
  if (t < NGRAPHS) cnt[t] = 0.0f;
}

__global__ void k_pool(const float* __restrict__ X, const int* __restrict__ batch,
                       float* __restrict__ pooled, float* __restrict__ cnt) {
  int i = blockIdx.x;           // node
  int f = threadIdx.x;          // feature (blockDim = 128)
  int g = batch[i];
  atomicAdd(&pooled[g * HID + f], X[(size_t)i * HID + f]);
  if (f == 0) atomicAdd(&cnt[g], 1.0f);
}

__global__ void k_head(const float* __restrict__ pooled, const float* __restrict__ cnt,
                       const float* __restrict__ Wp, const float* __restrict__ bp,
                       float* __restrict__ out) {
  int g = blockIdx.x * blockDim.x + threadIdx.x;
  if (g >= NGRAPHS) return;
  float inv = 1.0f / fmaxf(cnt[g], 1.0f);
  float logits[NCLS];
#pragma unroll
  for (int j = 0; j < NCLS; ++j) logits[j] = bp[j];
  for (int f = 0; f < HID; ++f) {
    float p = pooled[g * HID + f] * inv;
#pragma unroll
    for (int j = 0; j < NCLS; ++j) logits[j] = fmaf(p, Wp[f * NCLS + j], logits[j]);
  }
  float m = logits[0];
#pragma unroll
  for (int j = 1; j < NCLS; ++j) m = fmaxf(m, logits[j]);
  float sum = 0.0f;
#pragma unroll
  for (int j = 0; j < NCLS; ++j) sum += __expf(logits[j] - m);
  float lse = __logf(sum);
#pragma unroll
  for (int j = 0; j < NCLS; ++j) out[g * NCLS + j] = logits[j] - m - lse;
}

// ---------------- launch ----------------

extern "C" void kernel_launch(void* const* d_in, const int* in_sizes, int n_in,
                              void* d_out, int out_size, void* d_ws, size_t ws_size,
                              hipStream_t stream) {
  const int*   edge_index = (const int*)d_in[0];
  const int*   batch      = (const int*)d_in[1];
  const float* W0 = (const float*)d_in[2];
  const float* b0 = (const float*)d_in[3];
  const float* W1 = (const float*)d_in[4];
  const float* b1 = (const float*)d_in[5];
  const float* W2 = (const float*)d_in[6];
  const float* b2 = (const float*)d_in[7];
  const float* Wp = (const float*)d_in[8];
  const float* bp = (const float*)d_in[9];

  const int E = in_sizes[0] / 2;
  const int N = in_sizes[1];
  const int* src = edge_index;       // edge_index[0]
  const int* dst = edge_index + E;   // edge_index[1]
  const int total = N * HID;         // 12.8M, fits int
  const int nquad = total / 4;       // HID % 4 == 0
  const int fullTiles = N / 16;      // 6250 for N=100000
  const int tailRows  = N - fullTiles * 16;  // 0 for this problem

  // carve workspace (256B aligned)
  char* ws = (char*)d_ws;
  size_t off = 0;
  auto carve = [&](size_t bytes) -> char* {
    off = (off + 255) & ~(size_t)255;
    char* p = ws + off;
    off += bytes;
    return p;
  };
  float*    dinv   = (float*)carve((size_t)N * 4);
  float*    sacc   = (float*)carve((size_t)N * 4);
  float*    norm   = (float*)carve((size_t)E * 4);
  _Float16* x16    = (_Float16*)carve((size_t)total * 2);
  float*    h      = (float*)carve((size_t)total * 4);
  float*    agg    = (float*)carve((size_t)total * 4);
  _Float16* wt16   = (_Float16*)carve((size_t)HID * HID * 2);
  float*    pooled = (float*)carve((size_t)NGRAPHS * HID * 4);
  float*    cnt    = (float*)carve((size_t)NGRAPHS * 4);
  (void)ws_size; (void)n_in; (void)out_size;

  const int B = 256;
  dim3 blk(B);

  // norm setup
  k_fill1   <<<(N + B - 1) / B, blk, 0, stream>>>(dinv, N);
  k_deg     <<<(E + B - 1) / B, blk, 0, stream>>>(dst, dinv, E);
  k_rsqrt   <<<(N + B - 1) / B, blk, 0, stream>>>(dinv, N);
  k_selfnorm<<<(N + B - 1) / B, blk, 0, stream>>>(dinv, sacc, N);
  k_edgenorm<<<(E + B - 1) / B, blk, 0, stream>>>(src, dst, dinv, norm, sacc, E);

  // layer 0 (x == ones -> closed form)
  k_layer0<<<(nquad + B - 1) / B, blk, 0, stream>>>(sacc, W0, b0, x16, nquad);

  // layers 1 and 2
  const float* Ws[2] = {W1, W2};
  const float* bs[2] = {b1, b2};
  for (int l = 0; l < 2; ++l) {
    bool last = (l == 1);
    k_prepw<<<(HID * HID + B - 1) / B, blk, 0, stream>>>(Ws[l], wt16);
    if (fullTiles > 0)
      k_gemm<<<fullTiles, blk, 0, stream>>>(x16, wt16, h);
    if (tailRows > 0)
      k_gemm_tail<<<(tailRows * HID + B - 1) / B, blk, 0, stream>>>(
          x16, wt16, h, fullTiles * 16, N);
    k_agginit<<<(nquad + B - 1) / B, blk, 0, stream>>>(h, dinv, agg, nquad);
    k_scatter<<<(E + 7) / 8, blk, 0, stream>>>(src, dst, norm, h, agg, E);
    k_finish <<<(nquad + B - 1) / B, blk, 0, stream>>>(agg, bs[l], x16,
                                                       last ? h : (float*)nullptr, nquad);
  }

  // mean pool per graph + linear head + log_softmax
  k_zeropool<<<(NGRAPHS * HID + B - 1) / B, blk, 0, stream>>>(pooled, cnt);
  k_pool    <<<N, dim3(HID), 0, stream>>>(h, batch, pooled, cnt);
  k_head    <<<(NGRAPHS + B - 1) / B, blk, 0, stream>>>(pooled, cnt, Wp, bp, (float*)d_out);
}
